// NonLocalBlock2D_62002147885122
// MI455X (gfx1250) — matrix operand order repaired and checked by reading, not compiled, not run on hardware
//
#include <hip/hip_runtime.h>
#include <stdint.h>

// ---------------------------------------------------------------------------
// NonLocalBlock2D for MI455X (gfx1250): flash-attention with f16 WMMA and
// Tensor-Data-Mover (async, TENSORcnt-tracked) K/V staging with HW padding.
// B=8, C=256, D=128, N=4096.
// ---------------------------------------------------------------------------

typedef __attribute__((ext_vector_type(16))) _Float16 v16h;
typedef __attribute__((ext_vector_type(8)))  _Float16 v8h;
typedef __attribute__((ext_vector_type(8)))  float    v8f;
typedef __attribute__((ext_vector_type(4)))  unsigned int u32x4;
typedef __attribute__((ext_vector_type(8)))  int      i32x8;
typedef __attribute__((ext_vector_type(4)))  int      i32x4;

#define WMMA_F16(A, Bm, Cacc) \
  __builtin_amdgcn_wmma_f32_16x16x32_f16(false, (A), false, (Bm), (short)0, (Cacc), false, false)

#if __has_builtin(__builtin_amdgcn_tensor_load_to_lds) && \
    __has_builtin(__builtin_amdgcn_s_wait_tensorcnt)
#define USE_TDM 1
#endif

__device__ __forceinline__ v8f zero_v8f() {
  v8f z;
#pragma unroll
  for (int i = 0; i < 8; ++i) z[i] = 0.0f;
  return z;
}

__device__ __forceinline__ v16h pack_a(v8h lo, v8h hi) {
  // A-fragment: K linear per lane, 16 contiguous values
  v16h a;
#pragma unroll
  for (int e = 0; e < 8; ++e) { a[e] = lo[e]; a[e + 8] = hi[e]; }
  return a;
}

__device__ __forceinline__ v16h pack_b(v8h lo, v8h hi) {
  // B-fragment: K linear per lane, 16 contiguous values
  v16h b;
#pragma unroll
  for (int e = 0; e < 8; ++e) { b[e] = lo[e]; b[e + 8] = hi[e]; }
  return b;
}

#define NB  8
#define NC  256
#define ND  128
#define NN  4096

#ifdef USE_TDM
// ---------------------------------------------------------------------------
// TDM 2-D tile load: tile_d1 rows x tile_d0 elems (2-byte elements), row
// stride `stride0` elements in memory, written to LDS row-major with
// pad_amount(+1) DWORDs appended every 2^(pad_interval+1) DWORDs (= one row).
// D# packing per CDNA5 ISA ch.8 (group0: count/lds/global/type, group1: dims).
// Toolchain here exposes the 6-arg builtin:
//   (u32x4 g0, i32x8 g1, i32x4 g2, i32x4 g3, i32x8 extra, i32 cpol)
// groups 2/3/extra are zero for a 2-D tensor.
// ---------------------------------------------------------------------------
__device__ __forceinline__ void tdm_load_2d(uint32_t lds_addr, const void* gptr,
                                            uint32_t tensor_d0, uint32_t tensor_d1,
                                            uint64_t stride0,
                                            uint32_t tile_d0, uint32_t tile_d1,
                                            uint32_t pad_interval, uint32_t pad_amount)
{
  uint64_t ga = (uint64_t)(uintptr_t)gptr;
  u32x4 g0;
  g0[0] = 1u;                                                   // count=1, user mode
  g0[1] = (uint32_t)__builtin_amdgcn_readfirstlane((int)lds_addr);
  g0[2] = (uint32_t)__builtin_amdgcn_readfirstlane((int)(uint32_t)ga);
  g0[3] = ((uint32_t)__builtin_amdgcn_readfirstlane((int)(uint32_t)(ga >> 32)) & 0x01FFFFFFu)
          | 0x80000000u;                                        // type=2 ("image")
  i32x8 g1;
  g1[0] = (int)((1u << 16)                 // data_size = 2 bytes
              | (1u << 20)                 // pad_enable
              | (pad_interval << 22)
              | (pad_amount << 25));
  g1[1] = (int)((tensor_d0 & 0xFFFFu) << 16);                   // tensor_dim0[15:0]
  g1[2] = (int)((tensor_d0 >> 16) | ((tensor_d1 & 0xFFFFu) << 16));
  g1[3] = (int)((tensor_d1 >> 16) | (tile_d0 << 16));
  g1[4] = (int)tile_d1;                                         // tile_dim2 = 0
  g1[5] = (int)(uint32_t)stride0;                               // dim0 stride lo
  g1[6] = (int)((uint32_t)(stride0 >> 32) & 0xFFFFu);           // stride hi, dim1_stride=0
  g1[7] = 0;
  i32x4 z4 = {0, 0, 0, 0};                                      // groups 2/3: 2-D tensor
  i32x8 z8 = {0, 0, 0, 0, 0, 0, 0, 0};
  __builtin_amdgcn_tensor_load_to_lds(g0, g1, z4, z4, z8, 0);
}
#endif

// ---------------------------------------------------------------------------
// Kernel 1: projections. out[n][d] = sum_c x[c][n] * W[d][c] + bias[d]
// A = x^T tile (16n x 32c) via LDS transpose, B = W rows (32c x 16d) direct.
// theta/phi stored [B][N][D]; g stored transposed [B][D][N] for TDM staging.
// ---------------------------------------------------------------------------
__global__ __launch_bounds__(32)
void nlb_proj_kernel(const float* __restrict__ x,
                     const float* __restrict__ Wg, const float* __restrict__ bg,
                     const float* __restrict__ Wt, const float* __restrict__ bt,
                     const float* __restrict__ Wp, const float* __restrict__ bp,
                     _Float16* __restrict__ Th, _Float16* __restrict__ Ph,
                     _Float16* __restrict__ Gt)
{
  const int ntile = blockIdx.x;   // 0..255
  const int dtile = blockIdx.y;   // 0..7
  const int b     = blockIdx.z;   // 0..7
  const int lane  = threadIdx.x;
  const int l15   = lane & 15;
  const int hi    = lane >> 4;

  __shared__ _Float16 xT[16][40];   // [n_local][c_local], 80B row (16B aligned)

  const float* xb = x + (size_t)b * NC * NN + (size_t)ntile * 16;

  v8f accT = zero_v8f(), accP = zero_v8f(), accG = zero_v8f();

  const int dcol = dtile * 16 + l15;

  for (int k0 = 0; k0 < NC; k0 += 32) {
    // cooperative load x[k0..k0+31][16n] -> xT[n][c]  (coalesced global read)
#pragma unroll
    for (int it = 0; it < 16; ++it) {
      int c = it * 2 + hi;
      xT[l15][c] = (_Float16)xb[(size_t)(k0 + c) * NN + l15];
    }
    __syncthreads();

    // A fragment from LDS (two contiguous 16B reads per lane)
    const int kbA = hi * 16;
    v16h a = pack_a(*(const v8h*)&xT[l15][kbA],
                    *(const v8h*)&xT[l15][kbA + 8]);

    // B fragments direct from W rows (16 contiguous floats per lane)
    const int kbB = hi * 16;
    const float* wt = Wt + (size_t)dcol * NC + k0 + kbB;
    const float* wp = Wp + (size_t)dcol * NC + k0 + kbB;
    const float* wg = Wg + (size_t)dcol * NC + k0 + kbB;
    v16h bT, bP, bG;
#pragma unroll
    for (int e = 0; e < 16; ++e) {
      bT[e] = (_Float16)wt[e];
      bP[e] = (_Float16)wp[e];
      bG[e] = (_Float16)wg[e];
    }

    accT = WMMA_F16(a, bT, accT);
    accP = WMMA_F16(a, bP, accP);
    accG = WMMA_F16(a, bG, accG);
    __syncthreads();
  }

  const float biasT = bt[dcol], biasP = bp[dcol], biasG = bg[dcol];
#pragma unroll
  for (int r = 0; r < 8; ++r) {
    int n = ntile * 16 + r + hi * 8;          // C/D layout: M = r (+8 upper half)
    size_t off = ((size_t)b * NN + n) * ND + dcol;
    Th[off] = (_Float16)(accT[r] + biasT);
    Ph[off] = (_Float16)(accP[r] + biasP);
    Gt[((size_t)b * ND + dcol) * NN + n] = (_Float16)(accG[r] + biasG);
  }
}

// ---------------------------------------------------------------------------
// Kernel 2: flash attention. 4 waves/WG, wave owns 16 query rows (in regs).
// K/V tiles (32 keys) staged by TDM into double-buffered LDS with hardware
// padding (phi rows 256B+16B pad, gT rows 64B+16B pad), software-pipelined
// against the 16 WMMAs + online softmax of the previous tile.
// ---------------------------------------------------------------------------
__global__ __launch_bounds__(128)
void nlb_attn_kernel(const _Float16* __restrict__ Th,
                     const _Float16* __restrict__ Ph,
                     const _Float16* __restrict__ Gt,
                     _Float16* __restrict__ Yh)
{
  const int qblk = blockIdx.x;   // 0..63 -> 64 query rows per WG
  const int b    = blockIdx.y;
  const int tid  = threadIdx.x;
  const int wave = tid >> 5;
  const int lane = tid & 31;
  const int l15  = lane & 15;
  const int hi   = lane >> 4;

  __shared__ _Float16 phiT[2][32][136];     // [buf][m][d], 272B rows
  __shared__ _Float16 gT[2][128][40];       // [buf][d][m], 80B rows
  __shared__ _Float16 Pst[4][16][40];       // per-wave P staging [n][m]

  const int n0 = qblk * 64 + wave * 16;

  // Q fragments: theta rows n0..n0+15, K = d in 4 chunks of 32
  v16h qa[4];
  {
    const _Float16* qrow = Th + ((size_t)b * NN + n0 + l15) * ND;
    const int kbA = hi * 16;
#pragma unroll
    for (int dk = 0; dk < 4; ++dk)
      qa[dk] = pack_a(*(const v8h*)(qrow + dk * 32 + kbA),
                      *(const v8h*)(qrow + dk * 32 + kbA + 8));
  }

  v8f yacc[8];
#pragma unroll
  for (int dc = 0; dc < 8; ++dc) yacc[dc] = zero_v8f();
  float M[8], L[8];
#pragma unroll
  for (int r = 0; r < 8; ++r) { M[r] = -1e30f; L[r] = 0.0f; }

  const _Float16* Phb = Ph + (size_t)b * NN * ND;
  const _Float16* Gtb = Gt + (size_t)b * ND * NN;
  const int NT = NN / 32;                    // 128 key tiles

#ifdef USE_TDM
  if (tid < 32) {                            // TDM issued once per WG (EXEC-agnostic)
    tdm_load_2d((uint32_t)(uintptr_t)&phiT[0][0][0], Phb,
                ND, NN, ND, /*tile*/ ND, 32, /*pad: 64 dw*/ 5, /*4 dw*/ 3);
    tdm_load_2d((uint32_t)(uintptr_t)&gT[0][0][0], Gtb,
                NN, ND, NN, /*tile*/ 32, ND, /*pad: 16 dw*/ 3, /*4 dw*/ 3);
  }
#else
  const int cm = tid >> 2;                   // fallback cooperative staging
  const int cd = (tid & 3) * 32;
#endif

  for (int it = 0; it < NT; ++it) {
    const int buf = it & 1;
    const int m0 = it * 32;

#ifdef USE_TDM
    if (tid < 32) {
      if (it + 1 < NT) {                     // prefetch next tile into other buffer
        tdm_load_2d((uint32_t)(uintptr_t)&phiT[buf ^ 1][0][0],
                    Phb + (size_t)(m0 + 32) * ND,
                    ND, NN, ND, ND, 32, 5, 3);
        tdm_load_2d((uint32_t)(uintptr_t)&gT[buf ^ 1][0][0],
                    Gtb + (m0 + 32),
                    NN, ND, NN, 32, ND, 3, 3);
        __builtin_amdgcn_s_wait_tensorcnt(2);   // tile `it` complete (in-order)
      } else {
        __builtin_amdgcn_s_wait_tensorcnt(0);
      }
    }
    if (it + 2 < NT && tid >= 96)            // warm GL2 ahead of the TDM
      __builtin_prefetch(Phb + (size_t)(m0 + 64 + (tid - 96)) * ND, 0, 0);
    __syncthreads();
#else
    __syncthreads();
    {
      const _Float16* ps = Phb + (size_t)(m0 + cm) * ND + cd;
      v8h* pd = (v8h*)&phiT[buf][cm][cd];
#pragma unroll
      for (int i = 0; i < 4; ++i) pd[i] = *(const v8h*)(ps + i * 8);
      const _Float16* gs = Gtb + (size_t)tid * NN + m0;
      v8h* gd = (v8h*)&gT[buf][tid][0];
#pragma unroll
      for (int i = 0; i < 4; ++i) gd[i] = *(const v8h*)(gs + i * 8);
    }
    __syncthreads();
#endif

    // ---- scores: S0 (m cols 0..15), S1 (m cols 16..31) ----
    v8f s0 = zero_v8f(), s1 = zero_v8f();
    const int kbB = hi * 16;
#pragma unroll
    for (int dk = 0; dk < 4; ++dk) {
      const _Float16* p0 = &phiT[buf][l15][dk * 32 + kbB];
      const _Float16* p1 = &phiT[buf][16 + l15][dk * 32 + kbB];
      v16h b0 = pack_b(*(const v8h*)p0, *(const v8h*)(p0 + 8));
      v16h b1 = pack_b(*(const v8h*)p1, *(const v8h*)(p1 + 8));
      s0 = WMMA_F16(qa[dk], b0, s0);
      s1 = WMMA_F16(qa[dk], b1, s1);
    }

    // ---- online softmax, write P tile to wave-private LDS ----
#pragma unroll
    for (int r = 0; r < 8; ++r) {
      float a0 = s0[r], a1 = s1[r];
      float mx = fmaxf(a0, a1);
#pragma unroll
      for (int w = 8; w >= 1; w >>= 1) mx = fmaxf(mx, __shfl_xor(mx, w, 16));
      float mnew  = fmaxf(M[r], mx);
      float scale = __expf(M[r] - mnew);
      float p0 = __expf(a0 - mnew);
      float p1 = __expf(a1 - mnew);
      float sum = p0 + p1;
#pragma unroll
      for (int w = 8; w >= 1; w >>= 1) sum += __shfl_xor(sum, w, 16);
      L[r] = L[r] * scale + sum;
      M[r] = mnew;
#pragma unroll
      for (int dc = 0; dc < 8; ++dc) yacc[dc][r] *= scale;
      int nrow = r + hi * 8;                       // C/D layout row
      Pst[wave][nrow][l15]      = (_Float16)p0;
      Pst[wave][nrow][16 + l15] = (_Float16)p1;
    }

    // cross-lane LDS RAW inside one wave: drain DS before the re-layout read
#if __has_builtin(__builtin_amdgcn_s_wait_dscnt)
    __builtin_amdgcn_s_wait_dscnt(0);
#else
    asm volatile("s_wait_dscnt 0" ::: "memory");
#endif

    // re-read P as A-fragment (16n x 32m)
    const int kbA = hi * 16;
    const _Float16* pr = &Pst[wave][l15][0];
    v16h pa = pack_a(*(const v8h*)(pr + kbA),
                     *(const v8h*)(pr + kbA + 8));

    // ---- y += P @ g : 8 d-chunks of 16 ----
#pragma unroll
    for (int dc = 0; dc < 8; ++dc) {
      const _Float16* gp = &gT[buf][dc * 16 + l15][kbB];
      v16h bgf = pack_b(*(const v8h*)gp, *(const v8h*)(gp + 8));
      yacc[dc] = WMMA_F16(pa, bgf, yacc[dc]);
    }
    __syncthreads();            // all waves done reading buf before TDM reuses it
  }

  // ---- finalize: y /= L, store [B][N][D] f16 ----
#pragma unroll
  for (int r = 0; r < 8; ++r) {
    float inv = 1.0f / L[r];
    int n = n0 + r + hi * 8;
    _Float16* yrow = Yh + ((size_t)b * NN + n) * ND;
#pragma unroll
    for (int dc = 0; dc < 8; ++dc)
      yrow[dc * 16 + l15] = (_Float16)(yacc[dc][r] * inv);
  }
}

// ---------------------------------------------------------------------------
// Kernel 3: out[b][c][n] = sum_d Wm[c][d] * y[n][d] + bm[c] + x[b][c][n]
// A = Wm rows (16c x 32d) direct, B = y^T (contiguous per lane from [N][D]).
// ---------------------------------------------------------------------------
__global__ __launch_bounds__(32)
void nlb_outproj_kernel(const _Float16* __restrict__ Yh,
                        const float* __restrict__ Wm, const float* __restrict__ bm,
                        const float* __restrict__ x, float* __restrict__ out)
{
  const int ntile = blockIdx.x;   // 0..255
  const int ctile = blockIdx.y;   // 0..15
  const int b     = blockIdx.z;
  const int lane  = threadIdx.x;
  const int l15   = lane & 15;
  const int hi    = lane >> 4;

  v8f acc = zero_v8f();
  const int kbA = hi * 16;
  const int kbB = hi * 16;

#pragma unroll
  for (int dk = 0; dk < 4; ++dk) {
    const float* wrow = Wm + (size_t)(ctile * 16 + l15) * ND + dk * 32 + kbA;
    v16h a;
#pragma unroll
    for (int e = 0; e < 8; ++e) {
      a[e]     = (_Float16)wrow[e];
      a[e + 8] = (_Float16)wrow[e + 8];
    }
    const _Float16* yp = Yh + ((size_t)b * NN + ntile * 16 + l15) * ND + dk * 32 + kbB;
    v16h bb = pack_b(*(const v8h*)yp, *(const v8h*)(yp + 8));
    acc = WMMA_F16(a, bb, acc);
  }

#pragma unroll
  for (int r = 0; r < 8; ++r) {
    int c = ctile * 16 + r + hi * 8;
    int n = ntile * 16 + l15;
    size_t off = ((size_t)b * NC + c) * NN + n;
    out[off] = acc[r] + bm[c] + x[off];
  }
}

// ---------------------------------------------------------------------------
extern "C" void kernel_launch(void* const* d_in, const int* in_sizes, int n_in,
                              void* d_out, int out_size, void* d_ws, size_t ws_size,
                              hipStream_t stream) {
  (void)in_sizes; (void)n_in; (void)out_size; (void)ws_size;
  const float* x  = (const float*)d_in[0];
  const float* Wg = (const float*)d_in[1];
  const float* bg = (const float*)d_in[2];
  const float* Wt = (const float*)d_in[3];
  const float* bt = (const float*)d_in[4];
  const float* Wp = (const float*)d_in[5];
  const float* bp = (const float*)d_in[6];
  const float* Wm = (const float*)d_in[7];
  const float* bm = (const float*)d_in[8];
  float* out = (float*)d_out;

  char* ws = (char*)d_ws;
  const size_t MB8 = (size_t)8 << 20;          // B*N*D*2 = 8 MB per tensor
  _Float16* Th = (_Float16*)(ws);
  _Float16* Ph = (_Float16*)(ws + MB8);
  _Float16* Gt = (_Float16*)(ws + 2 * MB8);
  _Float16* Yh = (_Float16*)(ws + 3 * MB8);

  nlb_proj_kernel<<<dim3(NN / 16, ND / 16, NB), 32, 0, stream>>>(
      x, Wg, bg, Wt, bt, Wp, bp, Th, Ph, Gt);
  nlb_attn_kernel<<<dim3(NN / 64, NB), 128, 0, stream>>>(Th, Ph, Gt, Yh);
  nlb_outproj_kernel<<<dim3(NN / 16, NC / 16, NB), 32, 0, stream>>>(
      Yh, Wm, bm, x, out);
}
